// Decoder_44461501449121
// MI455X (gfx1250) — matrix operand order repaired and checked by reading, not compiled, not run on hardware
//
#include <hip/hip_runtime.h>
#include <math.h>

// Problem constants (from reference): B=32, T=50, S=40, V=32000, D=512, H=1024
constexpr int NB   = 32;
constexpr int NT   = 50;
constexpr int NS   = 40;
constexpr int NV   = 32000;
constexpr int ND   = 512;
constexpr int NH   = 1024;
constexpr int N4H  = 4 * NH;      // 4096
constexpr int KGATE = ND + NH;    // 1536 (fused [x|h] GEMM)
constexpr int KHT   = 2 * NH;     // 2048 (fused [ctx|h] GEMM)
constexpr int APAD  = 8;          // bf16 elems of row padding in LDS (16B) -> bank spread

typedef __attribute__((ext_vector_type(16))) __bf16 v16bf;
typedef __attribute__((ext_vector_type(8)))  __bf16 v8bf;
typedef __attribute__((ext_vector_type(8)))  float  v8f;

// B-fragment loader from global (per CDNA5 ISA 7.12.2 16-bit B 32x16 layout):
// lane L holds col N=L%16; group g=L/16 holds K = [kb+8g, kb+8g+8) and [kb+16+8g, kb+16+8g+8)
__device__ __forceinline__ v16bf load_b_frag(const __bf16* __restrict__ W,
                                             int row, int K, int kb, int g) {
  const v8bf lo = *reinterpret_cast<const v8bf*>(W + (size_t)row * K + kb + g * 8);
  const v8bf hi = *reinterpret_cast<const v8bf*>(W + (size_t)row * K + kb + 16 + g * 8);
  v16bf r;
#pragma unroll
  for (int i = 0; i < 8; ++i) { r[i] = lo[i]; r[i + 8] = hi[i]; }
  return r;
}

// C[32,N] = epilogue( A[32,K] @ W[N,K]^T + bias )
// A is staged into LDS once per workgroup via async global->LDS DMA, then WMMA
// A-fragments come from ds_load_b128 while the weight stream uses global_load_b128.
// One wave owns a full 32(M) x 32(N) tile: 4 accumulators, A/B frags reused 2x.
// MODE 0: outf = acc + bias0[n] + bias1[n]          (gates)
// MODE 1: outh = bf16(tanh(acc + bias0[n]))         (hx_atten)
// MODE 2: outf = acc + bias0[n]                     (logits, ldo = T*V)
template <int MODE>
__global__ void gemm32_wmma_bf16(const __bf16* __restrict__ A,
                                 const __bf16* __restrict__ W,
                                 const float* __restrict__ bias0,
                                 const float* __restrict__ bias1,
                                 float* __restrict__ outf,
                                 __bf16* __restrict__ outh,
                                 int K, int ldo) {
  extern __shared__ __bf16 Alds[];          // 32 x (K + APAD) bf16
  const int tid  = threadIdx.x;
  const int lane = tid & 31;
  const int wave = blockIdx.x * (blockDim.x >> 5) + (tid >> 5);
  const int nb   = wave * 32;               // 32 output columns per wave
  const int g    = lane >> 4;
  const int l16  = lane & 15;
  const int LDA  = K + APAD;                // padded LDS row stride (elements)

  // ---- async stage A[32xK] into LDS (one copy per workgroup, ASYNCcnt) ----
  {
    const int row_chunks = K >> 3;          // 16B chunks per row
    const int total      = 32 * row_chunks; // multiple of blockDim (K % 64 == 0)
    for (int c = tid; c < total; c += blockDim.x) {
      const int row = c / row_chunks;
      const int rc  = c - row * row_chunks;
      const unsigned lds_addr =
          (unsigned)(uintptr_t)(&Alds[(size_t)row * LDA + rc * 8]);
      const __bf16* gsrc = A + (size_t)row * K + rc * 8;
      asm volatile("global_load_async_to_lds_b128 %0, %1, off"
                   :: "v"(lds_addr), "v"(gsrc)
                   : "memory");
    }
    asm volatile("s_wait_asynccnt 0x0" ::: "memory");
    __syncthreads();
  }

  v8f c00 = {}, c01 = {}, c10 = {}, c11 = {};
  for (int kb = 0; kb < K; kb += 32) {
    // keep the weight stream warm in GL2 (emits global_prefetch_b8)
    __builtin_prefetch(W + (size_t)(nb + l16) * K + kb + 32, 0, 0);
    // A 16x32 fragment (ISA layout): lane L row M=L%16; group g holds
    // K = [kb+8g, kb+8g+8) and [kb+16+8g, kb+16+8g+8) -- two 16B ds loads
    v16bf a0, a1;
    {
      const v8bf lo0 = *reinterpret_cast<const v8bf*>(&Alds[(size_t)l16 * LDA + kb + g * 8]);
      const v8bf hi0 = *reinterpret_cast<const v8bf*>(&Alds[(size_t)l16 * LDA + kb + 16 + g * 8]);
      const v8bf lo1 = *reinterpret_cast<const v8bf*>(&Alds[(size_t)(l16 + 16) * LDA + kb + g * 8]);
      const v8bf hi1 = *reinterpret_cast<const v8bf*>(&Alds[(size_t)(l16 + 16) * LDA + kb + 16 + g * 8]);
#pragma unroll
      for (int i = 0; i < 8; ++i) {
        a0[i] = lo0[i]; a0[i + 8] = hi0[i];
        a1[i] = lo1[i]; a1[i + 8] = hi1[i];
      }
    }
    const v16bf b0 = load_b_frag(W, nb + l16,      K, kb, g);
    const v16bf b1 = load_b_frag(W, nb + 16 + l16, K, kb, g);
    c00 = __builtin_amdgcn_wmma_f32_16x16x32_bf16(false, a0, false, b0, (short)0, c00, false, false);
    c01 = __builtin_amdgcn_wmma_f32_16x16x32_bf16(false, a0, false, b1, (short)0, c01, false, false);
    c10 = __builtin_amdgcn_wmma_f32_16x16x32_bf16(false, a1, false, b0, (short)0, c10, false, false);
    c11 = __builtin_amdgcn_wmma_f32_16x16x32_bf16(false, a1, false, b1, (short)0, c11, false, false);
  }

  // C/D layout: VGPR r, lanes0-15 -> M=r,N=lane; lanes16-31 -> M=8+r,N=lane-16
  const int n0 = nb + l16;
  const int n1 = nb + 16 + l16;
  const float bA0 = bias0 ? bias0[n0] : 0.f;
  const float bA1 = bias0 ? bias0[n1] : 0.f;
  const float bB0 = bias1 ? bias1[n0] : 0.f;
  const float bB1 = bias1 ? bias1[n1] : 0.f;
#pragma unroll
  for (int r = 0; r < 8; ++r) {
    const int m0 = g * 8 + r;
    const int m1 = 16 + g * 8 + r;
    const float v00 = c00[r] + bA0 + bB0;
    const float v01 = c01[r] + bA1 + bB1;
    const float v10 = c10[r] + bA0 + bB0;
    const float v11 = c11[r] + bA1 + bB1;
    if (MODE == 1) {
      outh[(size_t)m0 * ldo + n0] = (__bf16)tanhf(v00);
      outh[(size_t)m0 * ldo + n1] = (__bf16)tanhf(v01);
      outh[(size_t)m1 * ldo + n0] = (__bf16)tanhf(v10);
      outh[(size_t)m1 * ldo + n1] = (__bf16)tanhf(v11);
    } else {
      outf[(size_t)m0 * ldo + n0] = v00;
      outf[(size_t)m0 * ldo + n1] = v01;
      outf[(size_t)m1 * ldo + n0] = v10;
      outf[(size_t)m1 * ldo + n1] = v11;
    }
  }
}

// ------------------------- setup / pointwise kernels -------------------------

__global__ void build_Wg_kernel(const float* __restrict__ W_ih,
                                const float* __restrict__ W_hh,
                                __bf16* __restrict__ Wg) {
  const int i = blockIdx.x * blockDim.x + threadIdx.x;
  if (i >= N4H * KGATE) return;
  const int row = i / KGATE, col = i % KGATE;
  const float v = (col < ND) ? W_ih[(size_t)row * ND + col]
                             : W_hh[(size_t)row * NH + (col - ND)];
  Wg[i] = (__bf16)v;
}

__global__ void f32_to_bf16_kernel(const float* __restrict__ src,
                                   __bf16* __restrict__ dst, int n) {
  const int i = blockIdx.x * blockDim.x + threadIdx.x;
  if (i < n) dst[i] = (__bf16)src[i];
}

__global__ void init_state_kernel(const float* __restrict__ h0,
                                  const float* __restrict__ c0,
                                  float* __restrict__ hx, float* __restrict__ cx) {
  const int i = blockIdx.x * blockDim.x + threadIdx.x;
  if (i < NB * NH) { hx[i] = h0[i]; cx[i] = c0[i]; }
}

__global__ void build_Ag_kernel(const int* __restrict__ zh,
                                const float* __restrict__ emb,
                                const float* __restrict__ hx,
                                __bf16* __restrict__ Ag, int t) {
  const int i = blockIdx.x * blockDim.x + threadIdx.x;
  if (i >= NB * KGATE) return;
  const int b = i / KGATE, c = i % KGATE;
  const float v = (c < ND) ? emb[(size_t)zh[b * NT + t] * ND + c]
                           : hx[(size_t)b * NH + (c - ND)];
  Ag[i] = (__bf16)v;
}

__global__ void lstm_pointwise_kernel(const float* __restrict__ gates,
                                      float* __restrict__ hx, float* __restrict__ cx) {
  const int i = blockIdx.x * blockDim.x + threadIdx.x;   // NB*NH
  const int b = i >> 10;
  const int h = i & (NH - 1);
  const float* g = gates + (size_t)b * N4H;
  const float gi = g[h], gf = g[NH + h], gg = g[2 * NH + h], go = g[3 * NH + h];
  const float si = 1.f / (1.f + expf(-gi));
  const float sf = 1.f / (1.f + expf(-gf));
  const float so = 1.f / (1.f + expf(-go));
  const float c = sf * cx[i] + si * tanhf(gg);
  cx[i] = c;
  hx[i] = so * tanhf(c);
}

__global__ void attn_scores_kernel(const float* __restrict__ enc,
                                   const float* __restrict__ hx,
                                   float* __restrict__ scores) {
  const int bs = blockIdx.x;              // b*NS + s
  const int b = bs / NS, s = bs % NS;
  __shared__ float red[128];
  float acc = 0.f;
  for (int h = threadIdx.x; h < NH; h += 128)
    acc += enc[((size_t)b * NS + s) * NH + h] * hx[(size_t)b * NH + h];
  red[threadIdx.x] = acc;
  __syncthreads();
  for (int off = 64; off > 0; off >>= 1) {
    if (threadIdx.x < off) red[threadIdx.x] += red[threadIdx.x + off];
    __syncthreads();
  }
  if (threadIdx.x == 0) scores[bs] = red[0];
}

__global__ void attn_softmax_ctx_kernel(const float* __restrict__ scores,
                                        const float* __restrict__ enc,
                                        const float* __restrict__ hx,
                                        __bf16* __restrict__ Aht) {
  const int b = blockIdx.x;
  __shared__ float attn[NS];
  if (threadIdx.x == 0) {
    float mx = -3.402823466e+38f;
    for (int s = 0; s < NS; ++s) mx = fmaxf(mx, scores[b * NS + s]);
    float sum = 0.f;
    for (int s = 0; s < NS; ++s) { const float e = expf(scores[b * NS + s] - mx); attn[s] = e; sum += e; }
    const float inv = 1.f / sum;
    for (int s = 0; s < NS; ++s) attn[s] *= inv;
  }
  __syncthreads();
  for (int h = threadIdx.x; h < NH; h += blockDim.x) {
    float acc = 0.f;
    for (int s = 0; s < NS; ++s)
      acc += attn[s] * enc[((size_t)b * NS + s) * NH + h];
    Aht[(size_t)b * KHT + h]       = (__bf16)acc;                       // [ctx | hx]
    Aht[(size_t)b * KHT + NH + h]  = (__bf16)hx[(size_t)b * NH + h];
  }
}

__global__ void argmax_row_kernel(const float* __restrict__ logits,
                                  float* __restrict__ pred, int t) {
  const int b = blockIdx.x;
  const float* row = logits + ((size_t)b * NT + t) * NV;
  __shared__ float sv[256];
  __shared__ int   si[256];
  float best = -3.402823466e+38f; int bidx = 0;
  for (int v = threadIdx.x; v < NV; v += 256) {
    const float x = row[v];
    if (x > best) { best = x; bidx = v; }
  }
  sv[threadIdx.x] = best; si[threadIdx.x] = bidx;
  __syncthreads();
  for (int off = 128; off > 0; off >>= 1) {
    if (threadIdx.x < off) {
      const float ov = sv[threadIdx.x + off]; const int oi = si[threadIdx.x + off];
      if (ov > sv[threadIdx.x] || (ov == sv[threadIdx.x] && oi < si[threadIdx.x])) {
        sv[threadIdx.x] = ov; si[threadIdx.x] = oi;
      }
    }
    __syncthreads();
  }
  if (threadIdx.x == 0) pred[b * NT + t] = (float)si[0];
}

// ---------------------------------------------------------------------------

static inline size_t smem_for_K(int K) { return (size_t)32 * (K + APAD) * 2; }

extern "C" void kernel_launch(void* const* d_in, const int* in_sizes, int n_in,
                              void* d_out, int out_size, void* d_ws, size_t ws_size,
                              hipStream_t stream) {
  const int*   zh    = (const int*)  d_in[0];
  const float* enc   = (const float*)d_in[1];
  const float* h0    = (const float*)d_in[2];
  const float* c0    = (const float*)d_in[3];
  // d_in[4] entext_len, d_in[5] teacher_forcing_ratio: unused by the math
  const float* emb   = (const float*)d_in[6];
  const float* W_ih  = (const float*)d_in[7];
  const float* W_hh  = (const float*)d_in[8];
  const float* b_ih  = (const float*)d_in[9];
  const float* b_hh  = (const float*)d_in[10];
  const float* W_ht  = (const float*)d_in[11];
  const float* b_ht  = (const float*)d_in[12];
  const float* W_voc = (const float*)d_in[13];
  const float* b_voc = (const float*)d_in[14];

  float* out_logits = (float*)d_out;                        // [B,T,V]
  float* out_pred   = out_logits + (size_t)NB * NT * NV;    // [B,T] (as float)

  // workspace carve-out (all 256B aligned); total ~83.4 MB
  char* ws = (char*)d_ws;
  size_t cur = 0;
  auto carve = [&](size_t bytes) -> char* {
    char* p = ws + cur;
    cur += (bytes + 255) & ~(size_t)255;
    return p;
  };
  __bf16* Wg_bf   = (__bf16*)carve((size_t)N4H * KGATE * 2);  // fused [W_ih|W_hh]
  __bf16* Wht_bf  = (__bf16*)carve((size_t)NH * KHT * 2);
  __bf16* Wvoc_bf = (__bf16*)carve((size_t)NV * NH * 2);
  float*  hx      = (float*) carve((size_t)NB * NH * 4);
  float*  cx      = (float*) carve((size_t)NB * NH * 4);
  __bf16* Ag      = (__bf16*)carve((size_t)NB * KGATE * 2);
  float*  gates   = (float*) carve((size_t)NB * N4H * 4);
  float*  scores  = (float*) carve((size_t)NB * NS * 4);
  __bf16* Aht     = (__bf16*)carve((size_t)NB * KHT * 2);
  __bf16* Avoc    = (__bf16*)carve((size_t)NB * NH * 2);
  (void)in_sizes; (void)n_in; (void)out_size; (void)ws_size;

  // One-time (per launch) weight conversion to bf16 + state init
  build_Wg_kernel<<<(N4H * KGATE + 255) / 256, 256, 0, stream>>>(W_ih, W_hh, Wg_bf);
  f32_to_bf16_kernel<<<(NH * KHT + 255) / 256, 256, 0, stream>>>(W_ht, Wht_bf, NH * KHT);
  f32_to_bf16_kernel<<<(NV * NH + 255) / 256, 256, 0, stream>>>(W_voc, Wvoc_bf, NV * NH);
  init_state_kernel<<<(NB * NH + 255) / 256, 256, 0, stream>>>(h0, c0, hx, cx);

  for (int t = 0; t < NT; ++t) {
    build_Ag_kernel<<<(NB * KGATE + 255) / 256, 256, 0, stream>>>(zh, emb, hx, Ag, t);
    // gates = [x|h] @ [W_ih|W_hh]^T + b_ih + b_hh   (M=32,N=4096,K=1536)
    gemm32_wmma_bf16<0><<<N4H / 256, 256, smem_for_K(KGATE), stream>>>(
        Ag, Wg_bf, b_ih, b_hh, gates, nullptr, KGATE, N4H);
    lstm_pointwise_kernel<<<(NB * NH) / 256, 256, 0, stream>>>(gates, hx, cx);
    attn_scores_kernel<<<NB * NS, 128, 0, stream>>>(enc, hx, scores);
    attn_softmax_ctx_kernel<<<NB, 256, 0, stream>>>(scores, enc, hx, Aht);
    // hx_atten = tanh([ctx|h] @ W_ht^T + b_ht)       (M=32,N=1024,K=2048) -> bf16
    gemm32_wmma_bf16<1><<<NH / 256, 256, smem_for_K(KHT), stream>>>(
        Aht, Wht_bf, b_ht, nullptr, nullptr, Avoc, KHT, NH);
    // logits = hx_atten @ W_voc^T + b_voc            (M=32,N=32000,K=1024) -> fp32 d_out
    gemm32_wmma_bf16<2><<<NV / 256, 256, smem_for_K(NH), stream>>>(
        Avoc, Wvoc_bf, b_voc, nullptr, out_logits + (size_t)t * NV, nullptr, NH, NT * NV);
    argmax_row_kernel<<<NB, 256, 0, stream>>>(out_logits, out_pred, t);
  }
}
